// RoPEAttention_65893388255870
// MI455X (gfx1250) — matrix operand order repaired and checked by reading, not compiled, not run on hardware
//
#include <hip/hip_runtime.h>
#include <hip/hip_bf16.h>

#define SEQ    4096
#define HID    1024
#define NHEAD  16
#define DHEAD  64
#define PATCHN 4064
#define KT     64          // flash attention key-tile width

typedef __bf16 bf16;
typedef bf16 v16bf __attribute__((ext_vector_type(16)));
typedef bf16 v8bf  __attribute__((ext_vector_type(8)));
typedef float v8f  __attribute__((ext_vector_type(8)));

__device__ __forceinline__ v8f wmma_bf16(v16bf a, v16bf b, v8f c) {
  // D = A(16x32 bf16) * B(32x16 bf16) + C(16x16 f32)
  return __builtin_amdgcn_wmma_f32_16x16x32_bf16(false, a, false, b, (short)0, c, false, false);
}

__device__ __forceinline__ v8f zerov8() {
  v8f z;
#pragma unroll
  for (int i = 0; i < 8; ++i) z[i] = 0.f;
  return z;
}

// ---- DPP16 cross-lane helpers (replace ds_bpermute shuffles; VALU-only) ----
// DPP ctrl: 0xB1 = quad_perm([1,0,3,2]) (xor1), 0x4E = quad_perm([2,3,0,1])
// (xor2), 0x141 = ROW_HALF_MIRROR, 0x140 = ROW_MIRROR. Any butterfly family
// covering bits 0..3 is valid for commutative reductions over a 16-lane row.
__device__ __forceinline__ float dpp_bcast_f(float x, int ctrl_unused) { return x; }

#define DPP_STEP_MAX(v, ctrl)                                                   \
  {                                                                             \
    int _t = __builtin_amdgcn_update_dpp(__float_as_int(v), __float_as_int(v),  \
                                         (ctrl), 0xf, 0xf, true);               \
    (v) = fmaxf((v), __int_as_float(_t));                                       \
  }
#define DPP_STEP_ADD(v, ctrl)                                                   \
  {                                                                             \
    int _t = __builtin_amdgcn_update_dpp(__float_as_int(v), __float_as_int(v),  \
                                         (ctrl), 0xf, 0xf, true);               \
    (v) = (v) + __int_as_float(_t);                                             \
  }

__device__ __forceinline__ float rowmax16_dpp(float v) {
  DPP_STEP_MAX(v, 0xB1);
  DPP_STEP_MAX(v, 0x4E);
  DPP_STEP_MAX(v, 0x141);
  DPP_STEP_MAX(v, 0x140);
  return v;
}
__device__ __forceinline__ float rowsum16_dpp(float v) {
  DPP_STEP_ADD(v, 0xB1);
  DPP_STEP_ADD(v, 0x4E);
  DPP_STEP_ADD(v, 0x141);
  DPP_STEP_ADD(v, 0x140);
  return v;
}
__device__ __forceinline__ float swap_xor1_dpp(float v) {
  int t = __builtin_amdgcn_update_dpp(__float_as_int(v), __float_as_int(v),
                                      0xB1, 0xf, 0xf, true);
  return __int_as_float(t);
}

// A operand: 16x32 bf16, row-major source with 'stride' elements per row.
// Lane l: row = l&15, K = kb..kb+15 contiguous, kb = 16*(l>>4).
__device__ __forceinline__ v16bf load_a_16x32(const bf16* base, int stride, int lane) {
  const int row = lane & 15;
  const int kb  = (lane >> 4) << 4;
  const bf16* p = base + (size_t)row * stride + kb;
  v8bf lo = *(const v8bf*)p;
  v8bf hh = *(const v8bf*)(p + 8);
  v16bf r;
#pragma unroll
  for (int i = 0; i < 8; ++i) { r[i] = lo[i]; r[i + 8] = hh[i]; }
  return r;
}

// B operand: 32x16 bf16 where column n is contiguous in memory (source is
// B^T row-major with 'stride' elements per row). Lane l: col = l&15,
// K = 16*(l>>4) + 0..15 contiguous -> single 32B load.
__device__ __forceinline__ v16bf load_b_32x16(const bf16* colmaj, size_t stride, int lane) {
  const int nn = lane & 15;
  const int kb = (lane >> 4) << 4;
  return *(const v16bf*)(colmaj + (size_t)nn * stride + kb);
}

// ---------------------------------------------------------------------------
// f32 -> bf16 bulk convert (8 elems/thread, packed 16B stores)
// ---------------------------------------------------------------------------
__global__ __launch_bounds__(256) void cvt_f32_bf16(const float* __restrict__ in,
                                                    bf16* __restrict__ out, int n8) {
  int i = blockIdx.x * blockDim.x + threadIdx.x;
  if (i >= n8) return;
  const float* p = in + (size_t)i * 8;
  v8bf o;
#pragma unroll
  for (int k = 0; k < 8; ++k) o[k] = (bf16)p[k];
  *(v8bf*)(out + (size_t)i * 8) = o;
}

// ---------------------------------------------------------------------------
// GEMM: C[4096 x 1024] = A(bf16,row-major) @ W(bf16,[N][K] row-major)^T
// Block = 4 waves (128 thr). Wave tile: 32(M) x 64(N), K-step 32, 8 WMMA/step.
// MODE 0: Q  -> fused RMSnorm + RoPE + 1/8 scale, store bf16 [h][s][d]
// MODE 1: K  -> fused RMSnorm + RoPE,             store bf16 [h][s][d]
// MODE 2: V  -> store bf16 transposed Vt[h][d][s] (packed 16B stores)
// MODE 3: out-> store fp32 row-major [s][1024]
// ---------------------------------------------------------------------------
template <int MODE>
__global__ __launch_bounds__(128) void gemm_qkvo(
    const bf16* __restrict__ A, const bf16* __restrict__ W,
    bf16* __restrict__ outb, float* __restrict__ outf,
    const float* __restrict__ nw, const float* __restrict__ rc,
    const float* __restrict__ rs) {
  const int lane = threadIdx.x & 31;
  const int wave = threadIdx.x >> 5;
  const int row0 = blockIdx.x * 128 + wave * 32;
  const int col0 = blockIdx.y * 64;  // == head*64 for MODE 0..2
  const int K = HID;

  v8f acc[2][4];
#pragma unroll
  for (int g = 0; g < 2; ++g)
#pragma unroll
    for (int t = 0; t < 4; ++t) acc[g][t] = zerov8();

  const bf16* a0p = A + (size_t)row0 * K;
  const bf16* a1p = A + (size_t)(row0 + 16) * K;
  for (int kc = 0; kc < K; kc += 32) {
    v16bf a0 = load_a_16x32(a0p + kc, K, lane);
    v16bf a1 = load_a_16x32(a1p + kc, K, lane);
#pragma unroll
    for (int t = 0; t < 4; ++t) {
      v16bf b = load_b_32x16(W + (size_t)(col0 + t * 16) * K + kc, K, lane);
      acc[0][t] = wmma_bf16(a0, b, acc[0][t]);
      acc[1][t] = wmma_bf16(a1, b, acc[1][t]);
    }
  }

  const int n = lane & 15, hig = lane >> 4;

  if (MODE <= 1) {
    // Wave holds the COMPLETE 64-wide head dim for 32 seq rows:
    // RMS-norm (4 tiles x 16-lane DPP reduce), RoPE, scale, all in fp32.
    const int h = col0 >> 6;
    const float qscale = (MODE == 0) ? 0.125f : 1.0f;
#pragma unroll
    for (int g = 0; g < 2; ++g) {
#pragma unroll
      for (int e = 0; e < 8; ++e) {
        float v[4];
        float s2 = 0.f;
#pragma unroll
        for (int t = 0; t < 4; ++t) { v[t] = acc[g][t][e]; s2 += v[t] * v[t]; }
        s2 = rowsum16_dpp(s2);
        const float rn = rsqrtf(s2 * (1.0f / 64.0f) + 1e-6f);
#pragma unroll
        for (int t = 0; t < 4; ++t) v[t] = v[t] * rn * nw[t * 16 + n];
        const int srow = row0 + g * 16 + hig * 8 + e;
        if (row0 >= (SEQ - PATCHN)) {  // rope boundary (=32) is wave-uniform
          const int p = srow - (SEQ - PATCHN);
#pragma unroll
          for (int t = 0; t < 4; ++t) {
            const int ii = t * 8 + (n >> 1);
            const float c = rc[(size_t)p * 32 + ii];
            const float s = rs[(size_t)p * 32 + ii];
            const float other = swap_xor1_dpp(v[t]);  // rope pair partner
            v[t] = (n & 1) ? (other * s + v[t] * c)   // odd: x0*s + x1*c
                           : (v[t] * c - other * s);  // even: x0*c - x1*s
          }
        }
#pragma unroll
        for (int t = 0; t < 4; ++t)
          outb[((size_t)h * SEQ + srow) * DHEAD + t * 16 + n] = (bf16)(v[t] * qscale);
      }
    }
  } else if (MODE == 2) {
    // Transposed store: Vt[h][d][s]; per (g,t) one packed v8bf (8 consecutive s).
    const int h = col0 >> 6;
#pragma unroll
    for (int g = 0; g < 2; ++g)
#pragma unroll
      for (int t = 0; t < 4; ++t) {
        v8bf pk;
#pragma unroll
        for (int e = 0; e < 8; ++e) pk[e] = (bf16)acc[g][t][e];
        const int d = t * 16 + n;
        const int sb = row0 + g * 16 + hig * 8;
        *(v8bf*)(outb + ((size_t)(h * DHEAD + d)) * SEQ + sb) = pk;
      }
  } else {
#pragma unroll
    for (int g = 0; g < 2; ++g)
#pragma unroll
      for (int t = 0; t < 4; ++t)
#pragma unroll
        for (int e = 0; e < 8; ++e)
          outf[(size_t)(row0 + g * 16 + hig * 8 + e) * HID + col0 + t * 16 + n] =
              acc[g][t][e];
  }
}

// ---------------------------------------------------------------------------
// Flash attention, CDNA5 style:
//  - block = 4 waves, one head, 4 adjacent 16-row query tiles
//  - K/V tiles (64 keys) staged in LDS with global_load_async_to_lds_b128,
//    double-buffered on ASYNCcnt, shared by all 4 waves (4x less L2 traffic)
//  - per 64-key tile: 8 WMMA QK^T + 8 WMMA P@V + 2 WMMA ones-column (row-sum
//    of P folded into the matrix pipe)
//  - running row-max butterfly done with DPP16 (VALU), zero DS-unit traffic
// ---------------------------------------------------------------------------
__global__ __launch_bounds__(128) void flash_attn(
    const bf16* __restrict__ Q, const bf16* __restrict__ Km,
    const bf16* __restrict__ Vt, bf16* __restrict__ AO) {
  __shared__ bf16 Kl[2][64 * 72];   // [key][d], row stride 72 (144B, 16B-aligned)
  __shared__ bf16 Vl[2][64 * 72];   // [d][key-chunk s]
  __shared__ bf16 Pl[4][16 * 72];   // per-wave P staging (C-layout -> A-layout)

  const int tid = threadIdx.x;
  const int lane = tid & 31;
  const int wave = tid >> 5;
  const int h = blockIdx.y;
  const int s0 = (blockIdx.x * 4 + wave) * 16;
  const int n = lane & 15, hig = lane >> 4;

  const bf16* qbase = Q + ((size_t)h * SEQ + s0) * DHEAD;
  const v16bf qa0 = load_a_16x32(qbase, DHEAD, lane);       // d 0..31
  const v16bf qa1 = load_a_16x32(qbase + 32, DHEAD, lane);  // d 32..63

  v16bf onesb;
#pragma unroll
  for (int i = 0; i < 16; ++i) onesb[i] = (bf16)1.0f;

  v8f acc[4], accL = zerov8();
#pragma unroll
  for (int t = 0; t < 4; ++t) acc[t] = zerov8();
  float Mr[8];
#pragma unroll
  for (int e = 0; e < 8; ++e) Mr[e] = -3.0e38f;

  // cooperative async stage of one 64-key K/V tile into LDS buffer `buf`
  auto issue_tile = [&](int buf, int j) {
    const int r = tid >> 1;           // 0..63 : K key row / V d row
    const int c = (tid & 1) * 32;     // element offset 0 or 32 within row
    const bf16* gk = Km + ((size_t)h * SEQ + j + r) * DHEAD + c;
    const bf16* gv = Vt + ((size_t)(h * DHEAD + r)) * SEQ + j + c;
    const unsigned lk = (unsigned)(size_t)&Kl[buf][r * 72 + c];
    const unsigned lv = (unsigned)(size_t)&Vl[buf][r * 72 + c];
    // INST_OFFSET is added to BOTH the LDS and global addresses (ISA 15.18.3)
    asm volatile("global_load_async_to_lds_b128 %0, %1, off"           :: "v"(lk), "v"(gk) : "memory");
    asm volatile("global_load_async_to_lds_b128 %0, %1, off offset:16" :: "v"(lk), "v"(gk) : "memory");
    asm volatile("global_load_async_to_lds_b128 %0, %1, off offset:32" :: "v"(lk), "v"(gk) : "memory");
    asm volatile("global_load_async_to_lds_b128 %0, %1, off offset:48" :: "v"(lk), "v"(gk) : "memory");
    asm volatile("global_load_async_to_lds_b128 %0, %1, off"           :: "v"(lv), "v"(gv) : "memory");
    asm volatile("global_load_async_to_lds_b128 %0, %1, off offset:16" :: "v"(lv), "v"(gv) : "memory");
    asm volatile("global_load_async_to_lds_b128 %0, %1, off offset:32" :: "v"(lv), "v"(gv) : "memory");
    asm volatile("global_load_async_to_lds_b128 %0, %1, off offset:48" :: "v"(lv), "v"(gv) : "memory");
  };

  issue_tile(0, 0);
  int buf = 0;
  for (int j = 0; j < SEQ; j += KT) {
    if (j + KT < SEQ) {
      issue_tile(buf ^ 1, j + KT);                       // prefetch next tile
      asm volatile("s_wait_asynccnt 8" ::: "memory");    // current tile done
    } else {
      asm volatile("s_wait_asynccnt 0" ::: "memory");
    }
    __syncthreads();

    // ---- scores: 4 groups of 16 keys, d = 64 split in two 32-chunks
    v8f Sg[4];
#pragma unroll
    for (int g = 0; g < 4; ++g) {
      v16bf b0 = load_b_32x16(&Kl[buf][(g * 16) * 72], 72, lane);
      v16bf b1 = load_b_32x16(&Kl[buf][(g * 16) * 72 + 32], 72, lane);
      Sg[g] = wmma_bf16(qa0, b0, zerov8());
      Sg[g] = wmma_bf16(qa1, b1, Sg[g]);
    }

    // ---- online softmax: row-max via DPP16 butterfly (no DS unit ops)
#pragma unroll
    for (int e = 0; e < 8; ++e) {
      float mx = fmaxf(fmaxf(Sg[0][e], Sg[1][e]), fmaxf(Sg[2][e], Sg[3][e]));
      mx = rowmax16_dpp(mx);
      const float newM = fmaxf(Mr[e], mx);
      const float sc = __expf(Mr[e] - newM);
      Mr[e] = newM;
      accL[e] *= sc;
#pragma unroll
      for (int t = 0; t < 4; ++t) acc[t][e] *= sc;
      const int m = hig * 8 + e;
#pragma unroll
      for (int g = 0; g < 4; ++g)
        Pl[wave][m * 72 + g * 16 + n] = (bf16)__expf(Sg[g][e] - newM);
    }
    asm volatile("s_wait_dscnt 0" ::: "memory");  // same-wave LDS RAW on Pl

    // ---- reload P in A-layout (16x64 -> two 16x32 chunks)
    v16bf pa0 = load_a_16x32(&Pl[wave][0], 72, lane);
    v16bf pa1 = load_a_16x32(&Pl[wave][32], 72, lane);

    // ---- denominator via ones-column matmul: accL += rowsum(P)
    accL = wmma_bf16(pa0, onesb, accL);
    accL = wmma_bf16(pa1, onesb, accL);

    // ---- P @ V
#pragma unroll
    for (int t = 0; t < 4; ++t) {
      v16bf vb0 = load_b_32x16(&Vl[buf][(t * 16) * 72], 72, lane);
      v16bf vb1 = load_b_32x16(&Vl[buf][(t * 16) * 72 + 32], 72, lane);
      acc[t] = wmma_bf16(pa0, vb0, acc[t]);
      acc[t] = wmma_bf16(pa1, vb1, acc[t]);
    }
    __syncthreads();   // all waves done reading before next overwrite
    buf ^= 1;
  }

  // normalize and emit bf16 attention output, row-major [s][h*64+d]
#pragma unroll
  for (int e = 0; e < 8; ++e) {
    const float inv = 1.0f / accL[e];
    const int srow = s0 + hig * 8 + e;
#pragma unroll
    for (int t = 0; t < 4; ++t)
      AO[(size_t)srow * HID + h * DHEAD + t * 16 + n] = (bf16)(acc[t][e] * inv);
  }
}

// ---------------------------------------------------------------------------
extern "C" void kernel_launch(void* const* d_in, const int* in_sizes, int n_in,
                              void* d_out, int out_size, void* d_ws, size_t ws_size,
                              hipStream_t stream) {
  (void)in_sizes; (void)n_in; (void)out_size; (void)ws_size;
  const float* x   = (const float*)d_in[0];
  const float* rc  = (const float*)d_in[1];
  const float* rs  = (const float*)d_in[2];
  const float* wq  = (const float*)d_in[3];
  const float* wk  = (const float*)d_in[4];
  const float* wv  = (const float*)d_in[5];
  const float* wo  = (const float*)d_in[6];
  const float* qnw = (const float*)d_in[7];
  const float* knw = (const float*)d_in[8];
  float* out = (float*)d_out;

  char* ws = (char*)d_ws;
  size_t off = 0;
  auto carve = [&](size_t bytes) -> void* {
    void* p = ws + off;
    off = (off + bytes + 255) & ~(size_t)255;
    return p;
  };
  bf16* xb  = (bf16*)carve((size_t)SEQ * HID * 2);
  bf16* wqb = (bf16*)carve((size_t)HID * HID * 2);
  bf16* wkb = (bf16*)carve((size_t)HID * HID * 2);
  bf16* wvb = (bf16*)carve((size_t)HID * HID * 2);
  bf16* wob = (bf16*)carve((size_t)HID * HID * 2);
  bf16* Qb  = (bf16*)carve((size_t)SEQ * HID * 2);  // [h][s][d]
  bf16* Kb  = (bf16*)carve((size_t)SEQ * HID * 2);  // [h][s][d]
  bf16* Vt  = (bf16*)carve((size_t)SEQ * HID * 2);  // [h][d][s]
  bf16* AO  = (bf16*)carve((size_t)SEQ * HID * 2);  // [s][h*64+d]

  const int n8x = SEQ * HID / 8;
  const int n8w = HID * HID / 8;
  cvt_f32_bf16<<<(n8x + 255) / 256, 256, 0, stream>>>(x, xb, n8x);
  cvt_f32_bf16<<<(n8w + 255) / 256, 256, 0, stream>>>(wq, wqb, n8w);
  cvt_f32_bf16<<<(n8w + 255) / 256, 256, 0, stream>>>(wk, wkb, n8w);
  cvt_f32_bf16<<<(n8w + 255) / 256, 256, 0, stream>>>(wv, wvb, n8w);
  cvt_f32_bf16<<<(n8w + 255) / 256, 256, 0, stream>>>(wo, wob, n8w);

  dim3 gg(SEQ / 128, HID / 64);
  gemm_qkvo<0><<<gg, 128, 0, stream>>>(xb, wqb, Qb, nullptr, qnw, rc, rs);
  gemm_qkvo<1><<<gg, 128, 0, stream>>>(xb, wkb, Kb, nullptr, knw, rc, rs);
  gemm_qkvo<2><<<gg, 128, 0, stream>>>(xb, wvb, Vt, nullptr, nullptr, nullptr, nullptr);

  flash_attn<<<dim3(SEQ / 64, NHEAD), 128, 0, stream>>>(Qb, Kb, Vt, AO);

  gemm_qkvo<3><<<gg, 128, 0, stream>>>(AO, wob, nullptr, out, nullptr, nullptr, nullptr);
}
